// WordsLstm_9277129359706
// MI455X (gfx1250) — matrix operand order repaired and checked by reading, not compiled, not run on hardware
//
#include <hip/hip_runtime.h>
#include <hip/hip_bf16.h>

typedef __attribute__((ext_vector_type(16))) __bf16 v16bf;
typedef __attribute__((ext_vector_type(8)))  __bf16 v8bf;
typedef __attribute__((ext_vector_type(4)))  __bf16 v4bf;
typedef __attribute__((ext_vector_type(8)))  float  v8f;

#define S_LEN 4096
#define HID   2048
#define IN_D  2048
#define NG    (4 * HID)   // 8192
#define SCAN_BLOCKS 8

// LDS tile geometry for the GEMM (per 128x128 block tile, K-step 32)
#define TILE_PITCH 80      // 64B of bf16 data + 16B pad (bank spread)
#define B_OFF      (128 * TILE_PITCH)      // 10240: B tile after A tile
#define BUF_STRIDE (2 * 128 * TILE_PITCH)  // 20480 per double-buffer slot

// ------------------------------------------------------------------
// fp32 -> bf16 elementwise convert (grid-stride)
// ------------------------------------------------------------------
__global__ void cvt_bf16_kernel(const float* __restrict__ in,
                                __bf16* __restrict__ out, long n) {
  long i = (long)blockIdx.x * blockDim.x + threadIdx.x;
  long stride = (long)gridDim.x * blockDim.x;
  for (; i < n; i += stride) out[i] = (__bf16)in[i];
}

// ------------------------------------------------------------------
// W_hh [NG][HID] fp32 -> wt [HID][NG] bf16 (transpose+convert)
// ------------------------------------------------------------------
__global__ void transpose_cvt_kernel(const float* __restrict__ w,
                                     __bf16* __restrict__ wt) {
  long i = (long)blockIdx.x * blockDim.x + threadIdx.x;
  long total = (long)NG * HID;
  long stride = (long)gridDim.x * blockDim.x;
  for (; i < total; i += stride) {
    long n = i / HID, k = i % HID;
    wt[k * (long)NG + n] = (__bf16)w[i];
  }
}

// ------------------------------------------------------------------
// init: copy h0/c0 into state buffers, zero the grid-sync area
// ------------------------------------------------------------------
__global__ void init_kernel(const float* __restrict__ h0, const float* __restrict__ c0,
                            float* __restrict__ h, float* __restrict__ c,
                            unsigned* __restrict__ sync) {
  int i = blockIdx.x * blockDim.x + threadIdx.x;
  if (i < HID) { h[i] = h0[i]; c[i] = c0[i]; }
  if (i < 64) sync[i] = 0u;
}

// ------------------------------------------------------------------
// CDNA5 async global->LDS copy of one K-slice (A 128x32 + B 128x32 bf16).
// Each of 256 threads supplies per-lane addresses for 2+2 B128 chunks;
// tracked by ASYNCcnt (no VGPR round-trip).
// ------------------------------------------------------------------
__device__ inline void issue_tile_async(const __bf16* __restrict__ A,
                                        const __bf16* __restrict__ B,
                                        int bm, int bn, int k,
                                        unsigned lds0, int tid) {
#pragma unroll
  for (int c = 0; c < 2; ++c) {
    int chunk = tid * 2 + c;          // 0..511
    int row   = chunk >> 2;           // 0..127
    int quad  = chunk & 3;            // which 16B of the 64B row
    const void* ga = (const void*)(A + (size_t)(bm + row) * IN_D + k + quad * 8);
    unsigned la = lds0 + row * TILE_PITCH + quad * 16;
    asm volatile("global_load_async_to_lds_b128 %0, %1, off"
                 :: "v"(la), "v"(ga) : "memory");
    const void* gb = (const void*)(B + (size_t)(bn + row) * IN_D + k + quad * 8);
    unsigned lb = lds0 + B_OFF + row * TILE_PITCH + quad * 16;
    asm volatile("global_load_async_to_lds_b128 %0, %1, off"
                 :: "v"(lb), "v"(gb) : "memory");
  }
}

__device__ inline void wait_async0() {
#if __has_builtin(__builtin_amdgcn_s_wait_asynccnt)
  __builtin_amdgcn_s_wait_asynccnt(0);
#else
  asm volatile("s_wait_asynccnt 0" ::: "memory");
#endif
}

// ------------------------------------------------------------------
// Phase 1: C[M=S_LEN, N=NG] = A[M,K] * B^T  (B = W_ih [NG, IN_D])
// bf16 WMMA 16x16x32, fp32 accumulate. Block 256 thr = 8 waves,
// block tile 128x128, wave tile 64(M)x32(N) = 4x2 accumulators.
// Double-buffered async global->LDS pipeline; fragments via ds loads.
// ------------------------------------------------------------------
__global__ __launch_bounds__(256) void gemm_wmma_kernel(
    const __bf16* __restrict__ A, const __bf16* __restrict__ B,
    float* __restrict__ C) {
  __shared__ __align__(128) unsigned char smem[2][BUF_STRIDE];

  const int tid  = threadIdx.x;
  const int lane = tid & 31;
  const int wv   = tid >> 5;                 // wave 0..7
  const int bm   = blockIdx.x * 128;
  const int bn   = blockIdx.y * 128;
  const int wmL  = (wv & 1) * 64;            // wave M origin in tile
  const int wnL  = (wv >> 1) * 32;           // wave N origin in tile
  const int half = lane >> 4;
  const int l16  = lane & 15;
  const unsigned sbase = (unsigned)(size_t)&smem[0][0];  // LDS byte offset

  v8f acc[4][2] = {};

  issue_tile_async(A, B, bm, bn, 0, sbase, tid);

  const int KS = IN_D / 32;                  // 64 K-steps
  for (int kk = 0; kk < KS; ++kk) {
    const int buf = kk & 1;
    wait_async0();                           // my wave's copies for tile kk done
    __syncthreads();                         // all waves' copies done; prev reads done
    if (kk + 1 < KS)
      issue_tile_async(A, B, bm, bn, (kk + 1) * 32,
                       sbase + (unsigned)((buf ^ 1) * BUF_STRIDE), tid);

    const unsigned char* sb = smem[buf];

    // A fragments (16x32 bf16): lanes 0-15 hold K 0-15, lanes 16-31 K 16-31
    v16bf af[4];
#pragma unroll
    for (int i = 0; i < 4; ++i) {
      const unsigned char* ap = sb + (wmL + 16 * i + l16) * TILE_PITCH;
      v8bf lo = *(const v8bf*)(ap + half * 32);
      v8bf hi = *(const v8bf*)(ap + 16 + half * 32);
      af[i] = __builtin_shufflevector(lo, hi, 0, 1, 2, 3, 4, 5, 6, 7,
                                              8, 9, 10, 11, 12, 13, 14, 15);
    }
    // B fragments (32x16): lane holds 16 contiguous K of column n
    v16bf bfg[2];
#pragma unroll
    for (int j = 0; j < 2; ++j) {
      const unsigned char* bp = sb + B_OFF + (wnL + 16 * j + l16) * TILE_PITCH + half * 32;
      bfg[j] = *(const v16bf*)bp;
    }
#pragma unroll
    for (int i = 0; i < 4; ++i)
#pragma unroll
      for (int j = 0; j < 2; ++j)
        acc[i][j] = __builtin_amdgcn_wmma_f32_16x16x32_bf16(
            false, af[i], false, bfg[j], (short)0, acc[i][j], false, false);
  }

  // C/D layout: VGPR v -> row = 16*i + v + 8*half, col = 16*j + l16
#pragma unroll
  for (int i = 0; i < 4; ++i)
#pragma unroll
    for (int j = 0; j < 2; ++j)
#pragma unroll
      for (int v = 0; v < 8; ++v) {
        int row = bm + wmL + 16 * i + v + 8 * half;
        int col = bn + wnL + 16 * j + l16;
        C[(size_t)row * NG + col] = acc[i][j][v];
      }
}

// ------------------------------------------------------------------
// device-wide barrier (sense via generation counter)
// ------------------------------------------------------------------
__device__ inline void grid_sync(unsigned* counter, volatile unsigned* gen, unsigned nb) {
  __syncthreads();
  if (threadIdx.x == 0) {
    __threadfence();                       // release
    unsigned g = *gen;
    if (atomicAdd(counter, 1u) == nb - 1) {
      *counter = 0u;
      __threadfence();
      atomicAdd((unsigned*)gen, 1u);
    } else {
      while (*gen == g) __builtin_amdgcn_s_sleep(1);
    }
    __threadfence();                       // acquire
  }
  __syncthreads();
}

// ------------------------------------------------------------------
// Phase 2: persistent recurrent scan.
// 8 blocks x 256 threads = 2048 threads; thread g owns gate outputs
// n0..n0+3 for the GEMV and h element g for the cell update.
// W_hh^T [HID][NG] bf16 stays L2-resident (32 MB of 192 MB L2);
// h staged into LDS each step for broadcast reads.
// ------------------------------------------------------------------
__global__ __launch_bounds__(256) void lstm_scan_kernel(
    const float* __restrict__ xg,     // [S_LEN, NG] precomputed x @ W_ih^T
    const __bf16* __restrict__ wt,    // [HID, NG]   W_hh^T bf16
    const float* __restrict__ bih, const float* __restrict__ bhh,
    float* h, float* c, float* gates, float* __restrict__ out,
    unsigned* sync) {
  __shared__ float sh[HID];
  const int gtid = blockIdx.x * 256 + threadIdx.x;   // 0..2047
  const int n0 = gtid * 4;
  unsigned* counter = sync;
  volatile unsigned* gen = sync + 16;

  for (int t = 0; t < S_LEN; ++t) {
    for (int k = threadIdx.x; k < HID; k += 256) sh[k] = h[k];
    __syncthreads();

    float a0 = 0.f, a1 = 0.f, a2 = 0.f, a3 = 0.f;
    const __bf16* wp = wt + n0;
#pragma unroll 8
    for (int k = 0; k < HID; ++k) {
      float hk = sh[k];
      v4bf w = *(const v4bf*)(wp + (size_t)k * NG);   // 8B coalesced per lane
      a0 += hk * (float)w[0];
      a1 += hk * (float)w[1];
      a2 += hk * (float)w[2];
      a3 += hk * (float)w[3];
    }
    const size_t xoff = (size_t)t * NG + n0;
    gates[n0 + 0] = xg[xoff + 0] + bih[n0 + 0] + bhh[n0 + 0] + a0;
    gates[n0 + 1] = xg[xoff + 1] + bih[n0 + 1] + bhh[n0 + 1] + a1;
    gates[n0 + 2] = xg[xoff + 2] + bih[n0 + 2] + bhh[n0 + 2] + a2;
    gates[n0 + 3] = xg[xoff + 3] + bih[n0 + 3] + bhh[n0 + 3] + a3;

    grid_sync(counter, gen, SCAN_BLOCKS);

    float gi = gates[gtid];
    float gf = gates[HID + gtid];
    float gg = gates[2 * HID + gtid];
    float go = gates[3 * HID + gtid];
    gi = 1.f / (1.f + __expf(-gi));
    gf = 1.f / (1.f + __expf(-gf));
    gg = tanhf(gg);
    go = 1.f / (1.f + __expf(-go));
    float cn = gf * c[gtid] + gi * gg;
    float hn = go * tanhf(cn);
    c[gtid] = cn;
    h[gtid] = hn;
    out[(size_t)t * HID + gtid] = hn;

    grid_sync(counter, gen, SCAN_BLOCKS);
  }
}

// ------------------------------------------------------------------
extern "C" void kernel_launch(void* const* d_in, const int* in_sizes, int n_in,
                              void* d_out, int out_size, void* d_ws, size_t ws_size,
                              hipStream_t stream) {
  (void)in_sizes; (void)n_in; (void)out_size; (void)ws_size;
  const float* x   = (const float*)d_in[0];
  const float* Wih = (const float*)d_in[1];
  const float* Whh = (const float*)d_in[2];
  const float* bih = (const float*)d_in[3];
  const float* bhh = (const float*)d_in[4];
  const float* h0  = (const float*)d_in[5];
  const float* c0  = (const float*)d_in[6];
  float* out = (float*)d_out;

  char* ws = (char*)d_ws;
  size_t off = 0;
  auto take = [&](size_t bytes) -> char* {
    off = (off + 255) & ~(size_t)255;
    char* p = ws + off;
    off += bytes;
    return p;
  };
  unsigned* syncbuf = (unsigned*)take(256);
  __bf16* xb   = (__bf16*)take((size_t)S_LEN * IN_D * 2);   // 16 MB
  __bf16* wihb = (__bf16*)take((size_t)NG * IN_D * 2);      // 32 MB
  __bf16* wtb  = (__bf16*)take((size_t)HID * NG * 2);       // 32 MB
  float*  xgf  = (float*)take((size_t)S_LEN * NG * 4);      // 128 MB
  float*  hbuf = (float*)take((size_t)HID * 4);
  float*  cbuf = (float*)take((size_t)HID * 4);
  float*  gbuf = (float*)take((size_t)NG * 4);

  init_kernel<<<8, 256, 0, stream>>>(h0, c0, hbuf, cbuf, syncbuf);
  cvt_bf16_kernel<<<2048, 256, 0, stream>>>(x, xb, (long)S_LEN * IN_D);
  cvt_bf16_kernel<<<2048, 256, 0, stream>>>(Wih, wihb, (long)NG * IN_D);
  transpose_cvt_kernel<<<2048, 256, 0, stream>>>(Whh, wtb);

  dim3 g(S_LEN / 128, NG / 128);
  gemm_wmma_kernel<<<g, 256, 0, stream>>>(xb, wihb, xgf);

  lstm_scan_kernel<<<SCAN_BLOCKS, 256, 0, stream>>>(
      xgf, wtb, bih, bhh, hbuf, cbuf, gbuf, out, syncbuf);
}
